// MultiHeadAttention_25417616458445
// MI455X (gfx1250) — compile-verified, matrix operand order checked by reading, not run on hardware
//
#include <hip/hip_runtime.h>
#include <math.h>

// ---------------------------------------------------------------------------
// MHA forward for B=4, S=2048, D=768, H=12, HD=64 on gfx1250 (MI455X).
// All matmuls via v_wmma_f32_16x16x32_bf16 (bf16 inputs, fp32 accumulate).
// Pipeline:
//   0) convert embeds + Wq/Wk/Wv/Wo fp32 -> bf16 (once)
//   1) fused QKV GEMM + RoPE   (64x64 tile per wave, 16 WMMA / K-step)
//   2) causal flash attention  (32 queries per wave, online softmax)
//   3) output projection + bias
// ---------------------------------------------------------------------------

#define BB 4
#define SS 2048
#define DD 768
#define HH 12
#define HD 64

typedef __attribute__((ext_vector_type(16))) __bf16 v16bf;
typedef __attribute__((ext_vector_type(8)))  __bf16 v8bf;
typedef __attribute__((ext_vector_type(8)))  float  v8f;
typedef __attribute__((ext_vector_type(4)))  float  v4f;

__device__ __forceinline__ v8f wmma_bf16(v16bf a, v16bf b, v8f c) {
  return __builtin_amdgcn_wmma_f32_16x16x32_bf16(false, a, false, b,
                                                 (short)0, c, false, false);
}

// A/B fragment loader from a bf16 row-major matrix.
// rp = &M[row][0]; koff already includes the half-wave +8.
// Elements 0..7 = k..k+7, elements 8..15 = k+16..k+23 (ISA 16-bit layout).
__device__ __forceinline__ v16bf frag_bf16(const __bf16* __restrict__ rp, int koff) {
  v8bf lo = *reinterpret_cast<const v8bf*>(rp + koff);
  v8bf hi = *reinterpret_cast<const v8bf*>(rp + koff + 16);
  v16bf f;
#pragma unroll
  for (int i = 0; i < 8; ++i) { f[i] = lo[i]; f[8 + i] = hi[i]; }
  return f;
}

// ---------------------------------------------------------------------------
// Kernel 0: fp32 -> bf16 elementwise convert (n must be a multiple of 8).
// ---------------------------------------------------------------------------
__global__ __launch_bounds__(256) void cvt_bf16_kernel(
    const float* __restrict__ in, __bf16* __restrict__ out, int n8)
{
  const int i = (blockIdx.x * 256 + threadIdx.x);
  if (i < n8) {
    const v4f a = *reinterpret_cast<const v4f*>(in + i * 8);
    const v4f b = *reinterpret_cast<const v4f*>(in + i * 8 + 4);
    v8bf o;
#pragma unroll
    for (int j = 0; j < 4; ++j) { o[j] = (__bf16)a[j]; o[4 + j] = (__bf16)b[j]; }
    *reinterpret_cast<v8bf*>(out + i * 8) = o;
  }
}

// ---------------------------------------------------------------------------
// Kernel 1: fused QKV projection + RoPE.
// One wave -> 64 rows (tokens) x one head (64 cols) of one of {Q,K,V}:
// 4 A-frags x 4 B-frags -> 16 WMMAs per K-step, 8 fragment loads.
// Q,K bf16 (B,H,S,64); V bf16 transposed (B,H,64,S).
// ---------------------------------------------------------------------------
__global__ __launch_bounds__(128) void qkv_rope_kernel(
    const __bf16* __restrict__ X,    // (B*S, D) bf16
    const __bf16* __restrict__ Wq, const __bf16* __restrict__ Wk,
    const __bf16* __restrict__ Wv,
    __bf16* __restrict__ Qo, __bf16* __restrict__ Ko, __bf16* __restrict__ Vt)
{
  const int lane = threadIdx.x & 31;
  const int wave = threadIdx.x >> 5;
  const int job  = blockIdx.x * 4 + wave;          // 128 * 12 * 3 jobs
  const int mt   = job / (HH * 3);                 // 64-row M tile
  const int rem  = job % (HH * 3);
  const int h    = rem / 3;
  const int mat  = rem % 3;
  const __bf16* W = (mat == 0) ? Wq : (mat == 1) ? Wk : Wv;

  const int l15     = lane & 15;
  const int halfsel = (lane >> 4) & 1;
  const int koff    = halfsel * 8;

  const __bf16* arow[4];
  const __bf16* wrow[4];
#pragma unroll
  for (int a = 0; a < 4; ++a)
    arow[a] = X + (size_t)(mt * 64 + a * 16 + l15) * DD;
#pragma unroll
  for (int t = 0; t < 4; ++t)
    wrow[t] = W + (size_t)(h * 64 + t * 16 + l15) * DD;

  v8f z = {};
  v8f acc[4][4];
#pragma unroll
  for (int a = 0; a < 4; ++a)
#pragma unroll
    for (int t = 0; t < 4; ++t) acc[a][t] = z;

  for (int kk = 0; kk < DD; kk += 32) {
    v16bf af[4];
#pragma unroll
    for (int a = 0; a < 4; ++a) af[a] = frag_bf16(arow[a], kk + koff);
#pragma unroll
    for (int t = 0; t < 4; ++t) {
      const v16bf bf = frag_bf16(wrow[t], kk + koff);
#pragma unroll
      for (int a = 0; a < 4; ++a) acc[a][t] = wmma_bf16(af[a], bf, acc[a][t]);
    }
  }

  if (mat < 2) {
    // RoPE: q'[d] = q[d]*cos(s*if[d%32]) + (d<32 ? -q[d+32] : q[d-32])*sin(...)
    // Column of tile t is d = t*16 + l15; partner is tile t^2, same lane/row.
    const float LN1E4_32 = 0.28782313662425572f;   // ln(10000)/32
    const float invf0 = __expf(-(float)(l15)      * LN1E4_32);
    const float invf1 = __expf(-(float)(l15 + 16) * LN1E4_32);
#pragma unroll
    for (int a = 0; a < 4; ++a)
#pragma unroll
      for (int r = 0; r < 8; ++r) {
        const int m = mt * 64 + a * 16 + r + halfsel * 8;
        const float s = (float)(m & (SS - 1));
        const float c0 = __cosf(s * invf0), s0 = __sinf(s * invf0);
        const float c1 = __cosf(s * invf1), s1 = __sinf(s * invf1);
        const float v0 = acc[a][0][r] * c0 - acc[a][2][r] * s0;
        const float v1 = acc[a][1][r] * c1 - acc[a][3][r] * s1;
        const float v2 = acc[a][2][r] * c0 + acc[a][0][r] * s0;
        const float v3 = acc[a][3][r] * c1 + acc[a][1][r] * s1;
        acc[a][0][r] = v0; acc[a][1][r] = v1; acc[a][2][r] = v2; acc[a][3][r] = v3;
      }

    __bf16* O = (mat == 0) ? Qo : Ko;
#pragma unroll
    for (int a = 0; a < 4; ++a)
#pragma unroll
      for (int r = 0; r < 8; ++r) {
        const int m = mt * 64 + a * 16 + r + halfsel * 8;
        const int b = m >> 11, s = m & (SS - 1);
        __bf16* orow = O + ((size_t)(b * HH + h) * SS + s) * HD;
#pragma unroll
        for (int t = 0; t < 4; ++t)
          orow[t * 16 + l15] = (__bf16)acc[a][t][r];
      }
  } else {
#pragma unroll
    for (int a = 0; a < 4; ++a)
#pragma unroll
      for (int r = 0; r < 8; ++r) {
        const int m = mt * 64 + a * 16 + r + halfsel * 8;
        const int b = m >> 11, s = m & (SS - 1);
#pragma unroll
        for (int t = 0; t < 4; ++t) {
          const int d = t * 16 + l15;
          Vt[((size_t)(b * HH + h) * HD + d) * SS + s] = (__bf16)acc[a][t][r];
        }
      }
  }
}

// ---------------------------------------------------------------------------
// Kernel 2: causal flash attention (online softmax).
// Block = 4 waves = 128 queries of one (b,h); wave = 32 queries x HD=64
// (two 16-row subtiles u that share every K / V fragment load).
// ---------------------------------------------------------------------------
__global__ __launch_bounds__(128) void flash_attn_kernel(
    const __bf16* __restrict__ Q,    // (B,H,S,64)
    const __bf16* __restrict__ K,    // (B,H,S,64)
    const __bf16* __restrict__ Vt,   // (B,H,64,S)
    __bf16* __restrict__ ctx)        // (B,S,D)
{
  __shared__ __bf16 Pbuf[4][32 * 64];   // per-wave P tile (C-layout -> A-layout relay)

  const int lane  = threadIdx.x & 31;
  const int wave  = threadIdx.x >> 5;
  const int qblk  = blockIdx.x % (SS / 128);
  const int bh    = blockIdx.x / (SS / 128);
  const int qbase = qblk * 128 + wave * 32;        // first of this wave's 32 queries

  const __bf16* Qb = Q  + (size_t)bh * SS * HD;
  const __bf16* Kb = K  + (size_t)bh * SS * HD;
  const __bf16* Vb = Vt + (size_t)bh * HD * SS;

  const int l15     = lane & 15;
  const int halfsel = (lane >> 4) & 1;
  const int koff    = halfsel * 8;

  // Q A-fragments: subtile u covers rows qbase + u*16 .. +15.
  v16bf qa[2][2];
#pragma unroll
  for (int u = 0; u < 2; ++u) {
    const __bf16* qrow = Qb + (size_t)(qbase + u * 16 + l15) * HD;
    qa[u][0] = frag_bf16(qrow, 0  + koff);
    qa[u][1] = frag_bf16(qrow, 32 + koff);
  }

  v8f z = {};
  v8f o[2][4];
  float m_i[2][8], l_i[2][8];
#pragma unroll
  for (int u = 0; u < 2; ++u)
#pragma unroll
    for (int r = 0; r < 8; ++r) {
      o[u][r & 3] = z; o[u][(r & 3)] = z;  // (filled properly below)
      m_i[u][r] = -INFINITY; l_i[u][r] = 0.0f;
    }
#pragma unroll
  for (int u = 0; u < 2; ++u)
#pragma unroll
    for (int t = 0; t < 4; ++t) o[u][t] = z;

  const int kend = qbase + 64 - (qbase & 32);      // q+32 rounded up to 64
  for (int kb = 0; kb < kend; kb += 64) {
    // ---- S = Q . K^T : 32 x 64 scores, K frags shared by both subtiles ----
    v8f sf[2][4];
#pragma unroll
    for (int u = 0; u < 2; ++u)
#pragma unroll
      for (int t = 0; t < 4; ++t) sf[u][t] = z;
#pragma unroll
    for (int t = 0; t < 4; ++t) {
      const __bf16* krow = Kb + (size_t)(kb + t * 16 + l15) * HD;
      const v16bf kb0 = frag_bf16(krow, 0  + koff);
      const v16bf kb1 = frag_bf16(krow, 32 + koff);
      if (kb + 64 < kend)                              // prefetch next key block
        __builtin_prefetch(krow + (size_t)64 * HD, 0, 1);
#pragma unroll
      for (int u = 0; u < 2; ++u) {
        sf[u][t] = wmma_bf16(qa[u][0], kb0, sf[u][t]);
        sf[u][t] = wmma_bf16(qa[u][1], kb1, sf[u][t]);
      }
    }

    // ---- softmax update per subtile ----
#pragma unroll
    for (int u = 0; u < 2; ++u) {
      const int q0 = qbase + u * 16;
      const bool diag = (kb + 64 > q0);
      float rmax[8];
#pragma unroll
      for (int r = 0; r < 8; ++r) rmax[r] = -INFINITY;
#pragma unroll
      for (int t = 0; t < 4; ++t) {
        const int kcol = kb + t * 16 + l15;
#pragma unroll
        for (int r = 0; r < 8; ++r) {
          float v = sf[u][t][r] * 0.125f;              // 1/sqrt(64)
          if (diag && (kcol > q0 + r + halfsel * 8)) v = -INFINITY;
          sf[u][t][r] = v;
          rmax[r] = fmaxf(rmax[r], v);
        }
      }
#pragma unroll
      for (int off = 1; off < 16; off <<= 1)           // stays inside each half
#pragma unroll
        for (int r = 0; r < 8; ++r)
          rmax[r] = fmaxf(rmax[r], __shfl_xor(rmax[r], off));

      float alpha[8];
#pragma unroll
      for (int r = 0; r < 8; ++r) {
        const float mnew = fmaxf(m_i[u][r], rmax[r]);
        alpha[r]  = __expf(m_i[u][r] - mnew);
        m_i[u][r] = mnew;
      }
      float rsum[8];
#pragma unroll
      for (int r = 0; r < 8; ++r) rsum[r] = 0.0f;
#pragma unroll
      for (int t = 0; t < 4; ++t)
#pragma unroll
        for (int r = 0; r < 8; ++r) {
          const float p = __expf(sf[u][t][r] - m_i[u][r]);
          sf[u][t][r] = p;
          rsum[r] += p;
        }
#pragma unroll
      for (int off = 1; off < 16; off <<= 1)
#pragma unroll
        for (int r = 0; r < 8; ++r)
          rsum[r] += __shfl_xor(rsum[r], off);
#pragma unroll
      for (int r = 0; r < 8; ++r) l_i[u][r] = l_i[u][r] * alpha[r] + rsum[r];
#pragma unroll
      for (int t = 0; t < 4; ++t)
#pragma unroll
        for (int r = 0; r < 8; ++r) o[u][t][r] *= alpha[r];

      // P -> LDS (C-layout row = r + 8*halfsel, col = l15) as bf16
      __bf16* P = Pbuf[wave];
#pragma unroll
      for (int t = 0; t < 4; ++t)
#pragma unroll
        for (int r = 0; r < 8; ++r)
          P[(u * 16 + r + halfsel * 8) * 64 + t * 16 + l15] = (__bf16)sf[u][t][r];
    }
    asm volatile("s_wait_dscnt 0" ::: "memory");       // wave-local LDS RAW fence

    v16bf pa[2][2];
#pragma unroll
    for (int u = 0; u < 2; ++u) {
      const __bf16* prow = Pbuf[wave] + (u * 16 + l15) * 64;
      pa[u][0] = frag_bf16(prow, 0  + koff);
      pa[u][1] = frag_bf16(prow, 32 + koff);
    }

    // ---- O += P . V : V frags shared by both subtiles ----
#pragma unroll
    for (int t = 0; t < 4; ++t) {
      const __bf16* vrow = Vb + (size_t)(t * 16 + l15) * SS + kb;
      const v16bf vb0 = frag_bf16(vrow, 0  + koff);
      const v16bf vb1 = frag_bf16(vrow, 32 + koff);
#pragma unroll
      for (int u = 0; u < 2; ++u) {
        o[u][t] = wmma_bf16(pa[u][0], vb0, o[u][t]);
        o[u][t] = wmma_bf16(pa[u][1], vb1, o[u][t]);
      }
    }
  }

  // Epilogue: O /= l, store ctx bf16 at (b, s, h*64 + d).
  const int b = bh / HH, h = bh % HH;
#pragma unroll
  for (int u = 0; u < 2; ++u)
#pragma unroll
    for (int r = 0; r < 8; ++r) {
      const float rl = 1.0f / l_i[u][r];
      const int s = qbase + u * 16 + r + halfsel * 8;
      __bf16* crow = ctx + ((size_t)(b * SS + s)) * DD + h * 64;
#pragma unroll
      for (int t = 0; t < 4; ++t)
        crow[t * 16 + l15] = (__bf16)(o[u][t][r] * rl);
    }
}

// ---------------------------------------------------------------------------
// Kernel 3: output projection  out = ctx . Wo^T + bo   (fp32 out)
// 64x64 tile per wave: 16 WMMAs / 8 fragment loads per K-step.
// ---------------------------------------------------------------------------
__global__ __launch_bounds__(128) void out_proj_kernel(
    const __bf16* __restrict__ ctx,  // (B*S, D) bf16
    const __bf16* __restrict__ Wo,   // (D, D) bf16
    const float*  __restrict__ bo,   // (D,)
    float* __restrict__ out)         // (B*S, D) fp32
{
  const int lane = threadIdx.x & 31;
  const int wave = threadIdx.x >> 5;
  const int job  = blockIdx.x * 4 + wave;   // 128 * 12 jobs
  const int mt   = job / HH;                // 64-row M tile
  const int ng   = job % HH;                // 64-wide N group

  const int l15     = lane & 15;
  const int halfsel = (lane >> 4) & 1;
  const int koff    = halfsel * 8;

  const __bf16* arow[4];
  const __bf16* wrow[4];
#pragma unroll
  for (int a = 0; a < 4; ++a)
    arow[a] = ctx + (size_t)(mt * 64 + a * 16 + l15) * DD;
#pragma unroll
  for (int t = 0; t < 4; ++t)
    wrow[t] = Wo + (size_t)(ng * 64 + t * 16 + l15) * DD;

  v8f z = {};
  v8f acc[4][4];
#pragma unroll
  for (int a = 0; a < 4; ++a)
#pragma unroll
    for (int t = 0; t < 4; ++t) acc[a][t] = z;

  for (int kk = 0; kk < DD; kk += 32) {
    v16bf af[4];
#pragma unroll
    for (int a = 0; a < 4; ++a) af[a] = frag_bf16(arow[a], kk + koff);
#pragma unroll
    for (int t = 0; t < 4; ++t) {
      const v16bf bf = frag_bf16(wrow[t], kk + koff);
#pragma unroll
      for (int a = 0; a < 4; ++a) acc[a][t] = wmma_bf16(af[a], bf, acc[a][t]);
    }
  }

#pragma unroll
  for (int t = 0; t < 4; ++t) {
    const int n = ng * 64 + t * 16 + l15;
    const float bias = bo[n];
#pragma unroll
    for (int a = 0; a < 4; ++a)
#pragma unroll
      for (int r = 0; r < 8; ++r) {
        const int m = mt * 64 + a * 16 + r + halfsel * 8;
        out[(size_t)m * DD + n] = acc[a][t][r] + bias;
      }
  }
}

// ---------------------------------------------------------------------------
extern "C" void kernel_launch(void* const* d_in, const int* in_sizes, int n_in,
                              void* d_out, int out_size, void* d_ws, size_t ws_size,
                              hipStream_t stream) {
  const float* X  = (const float*)d_in[0];
  const float* Wq = (const float*)d_in[1];
  const float* Wk = (const float*)d_in[2];
  const float* Wv = (const float*)d_in[3];
  const float* Wo = (const float*)d_in[4];
  const float* bo = (const float*)d_in[5];
  float* out = (float*)d_out;

  const size_t NX = (size_t)BB * SS * DD;        // 6291456 activation elems
  const size_t NW = (size_t)DD * DD;             // 589824 weight elems
  const size_t N1 = (size_t)BB * HH * SS * HD;   // == NX, per Q/K/Vt/ctx

  char* ws = (char*)d_ws;
  __bf16* xb  = (__bf16*)(ws);
  __bf16* wqb = (__bf16*)(ws + 2 * (NX));
  __bf16* wkb = (__bf16*)(ws + 2 * (NX + 1 * NW));
  __bf16* wvb = (__bf16*)(ws + 2 * (NX + 2 * NW));
  __bf16* wob = (__bf16*)(ws + 2 * (NX + 3 * NW));
  __bf16* q   = (__bf16*)(ws + 2 * (NX + 4 * NW));
  __bf16* k   = (__bf16*)(ws + 2 * (NX + 4 * NW + 1 * N1));
  __bf16* vt  = (__bf16*)(ws + 2 * (NX + 4 * NW + 2 * N1));
  __bf16* ctx = (__bf16*)(ws + 2 * (NX + 4 * NW + 3 * N1));

  // 0) one-time fp32 -> bf16 conversion (kills per-K-step v_cvt in GEMMs)
  cvt_bf16_kernel<<<dim3((int)(NX / 8 + 255) / 256), dim3(256), 0, stream>>>(X,  xb,  (int)(NX / 8));
  cvt_bf16_kernel<<<dim3((int)(NW / 8 + 255) / 256), dim3(256), 0, stream>>>(Wq, wqb, (int)(NW / 8));
  cvt_bf16_kernel<<<dim3((int)(NW / 8 + 255) / 256), dim3(256), 0, stream>>>(Wk, wkb, (int)(NW / 8));
  cvt_bf16_kernel<<<dim3((int)(NW / 8 + 255) / 256), dim3(256), 0, stream>>>(Wv, wvb, (int)(NW / 8));
  cvt_bf16_kernel<<<dim3((int)(NW / 8 + 255) / 256), dim3(256), 0, stream>>>(Wo, wob, (int)(NW / 8));

  // 1) QKV + RoPE: 128 M-tiles x 12 heads x 3 mats = 4608 wave-jobs, 4/block.
  qkv_rope_kernel<<<dim3(4608 / 4), dim3(128), 0, stream>>>(xb, wqb, wkb, wvb, q, k, vt);

  // 2) Flash attention: (B*H) x (S/128) = 768 blocks, 4 waves each.
  flash_attn_kernel<<<dim3(BB * HH * (SS / 128)), dim3(128), 0, stream>>>(q, k, vt, ctx);

  // 3) Output projection: 128 x 12 = 1536 wave-jobs, 4/block.
  out_proj_kernel<<<dim3((128 * HH) / 4), dim3(128), 0, stream>>>(ctx, wob, bo, out);
}
